// VQVAE_22900765622692
// MI455X (gfx1250) — matrix operand order repaired and checked by reading, not compiled, not run on hardware
//
#include <hip/hip_runtime.h>

// ---------------------------------------------------------------------------
// CDNA5 (gfx1250) VQ-VAE forward.  All contractions via v_wmma_f32_16x16x32_bf16
// (wave32, 16x16 f32 accumulators = 8 VGPRs), fp32 in/out, bf16 staged in LDS.
// Conv block tile: 64(Cout) x 128(pixels); each wave does 4 WMMAs per K-chunk
// reusing its A fragment (weights) across 4 N-subtiles.
// ---------------------------------------------------------------------------

typedef __bf16 bf16_t;
typedef __attribute__((ext_vector_type(16))) __bf16   v16bf;
typedef __attribute__((ext_vector_type(8)))  float    v8f;
typedef __attribute__((ext_vector_type(4)))  unsigned u32x4;

union Frag { u32x4 q[2]; v16bf v; };

// Native f32 -> bf16 (RNE) — lowers to v_cvt on gfx1250.
__device__ __forceinline__ bf16_t f2bf(float f) { return (bf16_t)f; }

__device__ __forceinline__ v8f wmma_bf16(v16bf a, v16bf b, v8f c) {
    // (neg_a, A, neg_b, B, c_mod, C, reuse_a, reuse_b)
    return __builtin_amdgcn_wmma_f32_16x16x32_bf16(false, a, false, b, (short)0, c,
                                                   false, false);
}

#define F_RELU_IN    1
#define F_RELU_OUT   2
#define F_TRANSPOSED 4

// ---------------------------------------------------------------------------
// Implicit-GEMM conv / convT.  Block = 256 thr (8 waves).  Waves in 4x2 grid:
// wave (mw,nw) computes rows [mw*16,+16) x cols [nw*64,+64) as 4 x (16x16).
// K-loop over taps (kh,kw) and ci chunks of 32 (bf16 WMMA K).  LDS rows use
// stride 40 (80B) to keep 16B alignment while shifting banks.
// ---------------------------------------------------------------------------
__global__ __launch_bounds__(256)
void conv_wmma(const float* __restrict__ in, const float* __restrict__ wgt,
               const float* __restrict__ bias, const float* __restrict__ resid,
               float* __restrict__ out,
               int N, int Cin, int Hin, int Win,
               int Cout, int KH, int KW, int stride, int pad,
               int Ho, int Wo, int flags)
{
    __shared__ alignas(16) bf16_t As[64][40];    // [cout 0..63 ][k 0..31]
    __shared__ alignas(16) bf16_t Bs[128][40];   // [pix  0..127][k 0..31]

    const int tid   = threadIdx.x;
    const int lane  = tid & 31;
    const int wave  = tid >> 5;
    const int mw    = wave >> 1;     // 0..3
    const int nw    = wave & 1;      // 0..1
    const int khalf = lane >> 4;     // 0..1
    const int r16   = lane & 15;

    const int HW      = Ho * Wo;
    const int HinWin  = Hin * Win;
    const int pixBase = blockIdx.x * 128;
    const int coutBase= blockIdx.y * 64;

    // This thread's fixed B pixel for LDS fills (all 128 pixels share n: HW%128==0)
    const int bpix = tid & 127;
    const int p    = pixBase + bpix;
    const int n    = p / HW;
    const int prem = p - n * HW;
    const int oh   = prem / Wo;
    const int ow   = prem - oh * Wo;

    const bool relu_in    = (flags & F_RELU_IN)    != 0;
    const bool relu_out   = (flags & F_RELU_OUT)   != 0;
    const bool transposed = (flags & F_TRANSPOSED) != 0;

    v8f acc[4];
    #pragma unroll
    for (int t = 0; t < 4; ++t) acc[t] = (v8f){0.f,0.f,0.f,0.f,0.f,0.f,0.f,0.f};

    const int  ciChunks = (Cin + 31) >> 5;
    const long inN = (long)n * Cin * HinWin;

    for (int kh = 0; kh < KH; ++kh)
    for (int kw = 0; kw < KW; ++kw) {
        bool valid; int ih, iw;
        if (!transposed) {
            ih = oh * stride - pad + kh;
            iw = ow * stride - pad + kw;
            valid = (ih >= 0) & (ih < Hin) & (iw >= 0) & (iw < Win);
        } else {
            // ConvTranspose2d gather form: ih = (oh + pad - kh)/stride
            int th = oh + pad - kh;
            int tw = ow + pad - kw;
            valid = (th >= 0) & (tw >= 0);
            if (valid) valid = ((th % stride) == 0) & ((tw % stride) == 0);
            ih = th / stride; iw = tw / stride;
            valid = valid & (ih < Hin) & (iw < Win);
        }
        const long inBase = inN + (valid ? (long)ih * Win + iw : 0);

        for (int cc = 0; cc < ciChunks; ++cc) {
            const int ciBase = cc << 5;
            __syncthreads();
            // --- A tile: weights -> bf16 LDS (64x32, 8 elems/thread) -------
            {
                const int m   = tid & 63;
                const int co  = coutBase + m;
                const int kk0 = tid >> 6;            // 0..3
                #pragma unroll
                for (int j = 0; j < 8; ++j) {
                    int kk = kk0 + (j << 2);
                    int ci = ciBase + kk;
                    float w = 0.f;
                    if (co < Cout && ci < Cin) {
                        long widx = transposed
                            ? ((((long)ci * Cout + co) * KH + kh) * KW + kw)   // (Cin,Cout,kh,kw)
                            : ((((long)co * Cin + ci) * KH + kh) * KW + kw);   // (Cout,Cin,kh,kw)
                        w = wgt[widx];
                    }
                    As[m][kk] = f2bf(w);
                }
            }
            // --- B tile: im2col gather -> bf16 LDS (128x32, 16 elems/thr) --
            {
                const int kk0 = tid >> 7;            // 0..1
                #pragma unroll
                for (int j = 0; j < 16; ++j) {
                    int kk = kk0 + (j << 1);
                    int ci = ciBase + kk;
                    float v = 0.f;
                    if (valid && ci < Cin) {
                        v = in[inBase + (long)ci * HinWin];
                        if (relu_in) v = fmaxf(v, 0.f);
                    }
                    Bs[bpix][kk] = f2bf(v);
                }
            }
            __syncthreads();
            // --- fragments per ISA 16-bit A(16x32)/B(32x16) lane layout ----
            Frag fa;
            const bf16_t* pa = &As[(mw << 4) + r16][khalf << 3];
            fa.q[0] = *(const u32x4*)pa;            // K 8*khalf .. +7
            fa.q[1] = *(const u32x4*)(pa + 16);     // K 16+8*khalf .. +7
            #pragma unroll
            for (int t = 0; t < 4; ++t) {
                Frag fb;
                const bf16_t* pb = &Bs[(nw << 6) + (t << 4) + r16][khalf << 3];
                fb.q[0] = *(const u32x4*)pb;        // K 8*khalf .. +7
                fb.q[1] = *(const u32x4*)(pb + 16); // K 16+8*khalf .. +7
                acc[t] = wmma_bf16(fa.v, fb.v, acc[t]);
            }
        }
    }

    // --- epilogue: c[r] -> (M = r + 8*khalf, N = lane&15) ------------------
    const int premBase = pixBase % HW;
    const int prem0    = premBase + (nw << 6) + r16;
    const long outN    = (long)n * Cout * HW;
    #pragma unroll
    for (int rr = 0; rr < 8; ++rr) {
        const int co = coutBase + (mw << 4) + (khalf << 3) + rr;
        if (co < Cout) {
            const float bv = bias[co];
            const long o0 = outN + (long)co * HW + prem0;
            #pragma unroll
            for (int t = 0; t < 4; ++t) {
                float v = acc[t][rr] + bv;
                if (resid) v += resid[o0 + 16 * t];
                if (relu_out) v = fmaxf(v, 0.f);
                out[o0 + 16 * t] = v;
            }
        }
    }
}

// ---------------------------------------------------------------------------
// VQ argmin:  score = 0.5*||e||^2 - z.e  (same argmin as full squared dist).
// One wave per 16 pixels; WMMA over code chunks of 16, K=256.
// ---------------------------------------------------------------------------
__global__ __launch_bounds__(32)
void vq_kernel(const float* __restrict__ ze, const float* __restrict__ emb,
               const float* __restrict__ ecorr,
               int* __restrict__ idxOut, float* __restrict__ counts, int HW)
{
    const int C = 256, K = 512;
    __shared__ alignas(16) bf16_t Zs[16][264];   // 16 pixels x 256 dims (pad 8)
    const int tid     = threadIdx.x;
    const int pixBase = blockIdx.x * 16;

    for (int e = tid; e < 16 * 256; e += 32) {
        int pix = e >> 8;
        int c   = e & 255;
        int p   = pixBase + pix;
        int n   = p / HW, rem = p - n * HW;
        Zs[pix][c] = f2bf(ze[(long)n * C * HW + (long)c * HW + rem]);
    }
    __syncthreads();

    const int khalf = tid >> 4, r16 = tid & 15;
    float bestv = 3.4e38f; int besti = 0;

    for (int cb = 0; cb < K; cb += 16) {
        v8f acc = {0.f,0.f,0.f,0.f,0.f,0.f,0.f,0.f};
        const float* er = emb + (long)(cb + r16) * C;   // A row = code cb+r16
        for (int kc = 0; kc < C; kc += 32) {
            v16bf av;
            #pragma unroll
            for (int i = 0; i < 8; ++i) av[i]     = f2bf(er[kc + (khalf << 3) + i]);
            #pragma unroll
            for (int i = 0; i < 8; ++i) av[8 + i] = f2bf(er[kc + 16 + (khalf << 3) + i]);
            Frag fb;
            const bf16_t* pz = &Zs[r16][kc + (khalf << 3)];
            fb.q[0] = *(const u32x4*)pz;
            fb.q[1] = *(const u32x4*)(pz + 16);
            acc = wmma_bf16(av, fb.v, acc);
        }
        #pragma unroll
        for (int rr = 0; rr < 8; ++rr) {
            int code = cb + (khalf << 3) + rr;
            float v = ecorr[code] - acc[rr];
            if (v < bestv || (v == bestv && code < besti)) { bestv = v; besti = code; }
        }
    }
    // lane L and L^16 hold complementary code halves for pixel L&15
    float ov = __shfl_xor(bestv, 16, 32);
    int   oi = __shfl_xor(besti, 16, 32);
    if (ov < bestv || (ov == bestv && oi < besti)) { bestv = ov; besti = oi; }
    if (tid < 16) {
        idxOut[pixBase + tid] = besti;
        atomicAdd(&counts[besti], 1.0f);
    }
}

__global__ void ecorr_kernel(const float* __restrict__ emb, float* __restrict__ ecorr) {
    int k = blockIdx.x * blockDim.x + threadIdx.x;
    if (k >= 512) return;
    const float* e = emb + (long)k * 256;
    float s = 0.f;
    for (int c = 0; c < 256; ++c) s += e[c] * e[c];
    ecorr[k] = 0.5f * s;
}

__global__ void zq_gather(const int* __restrict__ idx, const float* __restrict__ emb,
                          float* __restrict__ zq, int HW, int total) {
    const int C = 256;
    int t = blockIdx.x * blockDim.x + threadIdx.x;
    if (t >= total) return;
    int chw = C * HW;
    int n   = t / chw;
    int rem = t - n * chw;
    int c   = rem / HW;
    int hw  = rem - c * HW;
    zq[t] = emb[(long)idx[n * HW + hw] * C + c];
}

__global__ void usage_kernel(const float* __restrict__ counts, float* __restrict__ usage) {
    int k = blockIdx.x * blockDim.x + threadIdx.x;
    if (k < 512) usage[k] = counts[k] * (1.0f / 65536.0f);   // counts sum to N*H*W
}

// ---------------------------------------------------------------------------
extern "C" void kernel_launch(void* const* d_in, const int* in_sizes, int n_in,
                              void* d_out, int out_size, void* d_ws, size_t ws_size,
                              hipStream_t stream)
{
    (void)in_sizes; (void)n_in; (void)out_size; (void)ws_size;
    const float* x     = (const float*)d_in[0];
    const float* ew0   = (const float*)d_in[1];
    const float* eb0   = (const float*)d_in[2];
    const float* ew1   = (const float*)d_in[3];
    const float* eb1   = (const float*)d_in[4];
    const float* ew2   = (const float*)d_in[5];
    const float* eb2   = (const float*)d_in[6];
    const float* er1w1 = (const float*)d_in[7];
    const float* er1b1 = (const float*)d_in[8];
    const float* er1w2 = (const float*)d_in[9];
    const float* er1b2 = (const float*)d_in[10];
    const float* er2w1 = (const float*)d_in[11];
    const float* er2b1 = (const float*)d_in[12];
    const float* er2w2 = (const float*)d_in[13];
    const float* er2b2 = (const float*)d_in[14];
    const float* emb   = (const float*)d_in[15];
    const float* dw0   = (const float*)d_in[16];
    const float* db0   = (const float*)d_in[17];
    const float* dr1w1 = (const float*)d_in[18];
    const float* dr1b1 = (const float*)d_in[19];
    const float* dr1w2 = (const float*)d_in[20];
    const float* dr1b2 = (const float*)d_in[21];
    const float* dr2w1 = (const float*)d_in[22];
    const float* dr2b1 = (const float*)d_in[23];
    const float* dr2w2 = (const float*)d_in[24];
    const float* dr2b2 = (const float*)d_in[25];
    const float* tw0   = (const float*)d_in[26];
    const float* tb0   = (const float*)d_in[27];
    const float* tw1   = (const float*)d_in[28];
    const float* tb1   = (const float*)d_in[29];

    const long S1 = 67108864L;   // 16*256*128*128
    const long S2 = 16777216L;   // 16*256*64*64
    float* ws     = (float*)d_ws;
    float* h0     = ws;              // 128x128 stage (reused for decoder upsample)
    float* bufA   = ws + S1;
    float* bufB   = bufA + S2;
    float* bufC   = bufB + S2;
    float* bufD   = bufC + S2;
    float* smallp = bufD + S2;
    int*   idxb   = (int*)smallp;         // 65536 indices
    float* counts = smallp + 65536;       // 512
    float* ecorr  = counts + 512;         // 512

    float* xhat  = (float*)d_out;                  // 16*3*256*256
    float* ze    = xhat + 16L * 3 * 256 * 256;     // 16*256*64*64
    float* zq    = ze + S2;                        // 16*256*64*64
    float* usage = zq + S2;                        // 512

    hipMemsetAsync(counts, 0, 512 * sizeof(float), stream);
    ecorr_kernel<<<2, 256, 0, stream>>>(emb, ecorr);

    auto conv = [&](const float* in, const float* w, const float* b, const float* resid,
                    float* out, int N, int Cin, int Hin, int Win, int Cout,
                    int KH, int KW, int s, int p, int Ho, int Wo, int flags) {
        dim3 grid((unsigned)(((long)N * Ho * Wo) / 128), (unsigned)((Cout + 63) / 64));
        conv_wmma<<<grid, 256, 0, stream>>>(in, w, b, resid, out, N, Cin, Hin, Win,
                                            Cout, KH, KW, s, p, Ho, Wo, flags);
    };

    // ---- encoder ----
    conv(x,    ew0,   eb0,   nullptr, h0,   16,   3, 256, 256, 256, 4, 4, 2, 1, 128, 128, F_RELU_OUT);
    conv(h0,   ew1,   eb1,   nullptr, bufA, 16, 256, 128, 128, 256, 4, 4, 2, 1,  64,  64, F_RELU_OUT);
    conv(bufA, ew2,   eb2,   nullptr, bufB, 16, 256,  64,  64, 256, 3, 3, 1, 1,  64,  64, 0);
    conv(bufB, er1w1, er1b1, nullptr, bufC, 16, 256,  64,  64, 256, 3, 3, 1, 1,  64,  64, F_RELU_IN);
    conv(bufC, er1w2, er1b2, bufB,    bufD, 16, 256,  64,  64, 256, 1, 1, 1, 0,  64,  64, F_RELU_IN);
    conv(bufD, er2w1, er2b1, nullptr, bufC, 16, 256,  64,  64, 256, 3, 3, 1, 1,  64,  64, F_RELU_IN);
    conv(bufC, er2w2, er2b2, bufD,    ze,   16, 256,  64,  64, 256, 1, 1, 1, 0,  64,  64, F_RELU_IN);

    // ---- vector quantization (dec_in == zq in forward pass) ----
    vq_kernel<<<65536 / 16, 32, 0, stream>>>(ze, emb, ecorr, idxb, counts, 4096);
    usage_kernel<<<2, 256, 0, stream>>>(counts, usage);
    zq_gather<<<16777216 / 256, 256, 0, stream>>>(idxb, emb, zq, 4096, 16777216);

    // ---- decoder ----
    conv(zq,   dw0,   db0,   nullptr, bufA, 16, 256,  64,  64, 256, 3, 3, 1, 1,  64,  64, 0);
    conv(bufA, dr1w1, dr1b1, nullptr, bufC, 16, 256,  64,  64, 256, 3, 3, 1, 1,  64,  64, F_RELU_IN);
    conv(bufC, dr1w2, dr1b2, bufA,    bufB, 16, 256,  64,  64, 256, 1, 1, 1, 0,  64,  64, F_RELU_IN);
    conv(bufB, dr2w1, dr2b1, nullptr, bufC, 16, 256,  64,  64, 256, 3, 3, 1, 1,  64,  64, F_RELU_IN);
    conv(bufC, dr2w2, dr2b2, bufB,    bufD, 16, 256,  64,  64, 256, 1, 1, 1, 0,  64,  64, F_RELU_IN);
    conv(bufD, tw0,   tb0,   nullptr, h0,   16, 256,  64,  64, 256, 4, 4, 2, 1, 128, 128, F_RELU_OUT | F_TRANSPOSED);
    conv(h0,   tw1,   tb1,   nullptr, xhat, 16, 256, 128, 128,   3, 4, 4, 2, 1, 256, 256, F_TRANSPOSED);
}
